// BiLSTMCRF_25984552141285
// MI455X (gfx1250) — matrix operand order repaired and checked by reading, not compiled, not run on hardware
//
#include <hip/hip_runtime.h>
#include <hip/hip_bf16.h>

// ---------- types ----------
typedef unsigned short u16;
typedef __bf16 v16bf __attribute__((ext_vector_type(16)));
typedef float  v8f   __attribute__((ext_vector_type(8)));
typedef unsigned int u32x4 __attribute__((ext_vector_type(4)));
typedef int i32x4 __attribute__((ext_vector_type(4)));
typedef int i32x8 __attribute__((ext_vector_type(8)));

#define LSEQ 4096
#define DW 512
#define DI 128
#define DX 640      // DW+DI
#define H2 512
#define G4 2048     // 4*H2
#define NT 128      // tags

#if defined(__has_builtin)
#if __has_builtin(__builtin_amdgcn_tensor_load_to_lds)
#define HAVE_TDM 1
#endif
#if __has_builtin(__builtin_amdgcn_s_cluster_barrier)
#define HAVE_CLUSTER_BAR 1
#endif
#endif

// ---------- helpers ----------
__device__ __forceinline__ u16 f2bf(float x) {
  unsigned u = __float_as_uint(x);
  unsigned r = u + 0x7fffu + ((u >> 16) & 1u);   // RNE
  return (u16)(r >> 16);
}
// pack 4 floats -> 4 bf16 in a uint2
__device__ __forceinline__ uint2 pack4bf(float4 v) {
  uint2 r;
  r.x = (unsigned)f2bf(v.x) | ((unsigned)f2bf(v.y) << 16);
  r.y = (unsigned)f2bf(v.z) | ((unsigned)f2bf(v.w) << 16);
  return r;
}
__device__ __forceinline__ float bf_lo(unsigned w) { return __uint_as_float(w << 16); }
__device__ __forceinline__ float bf_hi(unsigned w) { return __uint_as_float(w & 0xffff0000u); }
__device__ __forceinline__ float sigmoidf_(float x) { return 1.0f / (1.0f + __expf(-x)); }

// ---------- 0. init: h/c state + sync counters ----------
__global__ void init_state_kernel(const float* h0, const float* c0,
                                  float* h_cur, float* c_cur, unsigned* sync) {
  int t = threadIdx.x;                    // 1024 threads
  if (t < 2 * H2) { h_cur[t] = h0[t]; c_cur[t] = c0[t]; }
  if (t < 8) sync[t] = 0u;
}

// ---------- 1. fp32 -> bf16 (vectorized: float4 in, 4x bf16 out) ----------
__global__ void f32_to_bf16_kernel(const float* __restrict__ in, u16* __restrict__ out, int n4) {
  const float4* in4 = (const float4*)in;
  uint2* out2 = (uint2*)out;
  for (int i = blockIdx.x * blockDim.x + threadIdx.x; i < n4; i += gridDim.x * blockDim.x)
    out2[i] = pack4bf(in4[i]);
}

// ---------- 2. embedding gather -> x_bf16[L][640] (one float4 per lane) ----------
__global__ void gather_embed_kernel(const int* __restrict__ sentence, const int* __restrict__ intent,
                                    const float* __restrict__ word_emb, const float* __restrict__ intent_emb,
                                    u16* __restrict__ xb) {
  const int t = blockIdx.x;
  const int tid = threadIdx.x;            // 128 threads
  const int w = sentence[t], it = intent[t];
  // word part: 128 lanes x float4 = 512 floats
  float4 wv = ((const float4*)(word_emb + (size_t)w * DW))[tid];
  *(uint2*)&xb[(size_t)t * DX + tid * 4] = pack4bf(wv);
  // intent part: 32 lanes x float4 = 128 floats
  if (tid < DI / 4) {
    float4 iv = ((const float4*)(intent_emb + (size_t)it * DI))[tid];
    *(uint2*)&xb[(size_t)t * DX + DW + tid * 4] = pack4bf(iv);
  }
}

// ---------- 3. WMMA GEMM: C[M,N] = A[M,K](bf16,row) * B[N,K](bf16,row)^T + bias ----------
// block = 256 threads (8 waves); block tile 128(M) x 64(N); K step 32.
// B staging: exactly one global_load_b128 + one ds_store_b128 per thread per K-chunk.
__global__ void gemm_bf16_wmma_kernel(const u16* __restrict__ A, const u16* __restrict__ B,
                                      const float* __restrict__ bias, float* __restrict__ C,
                                      int M, int N, int K) {
  __shared__ u16 Bs[64 * 32];            // N-tile 64 x K-chunk 32 (row-major n x k)
  const int tid  = threadIdx.x;
  const int lane = tid & 31;
  const int wave = tid >> 5;
  const int row0 = blockIdx.x * 128 + wave * 16;
  const int colB = blockIdx.y * 64;

  v8f acc[4];
  v8f zero = {0.f, 0.f, 0.f, 0.f, 0.f, 0.f, 0.f, 0.f};
#pragma unroll
  for (int c = 0; c < 4; ++c) acc[c] = zero;

  const int m  = lane & 15;
  const int ka = (lane >> 4) * 8;        // A: lanes<16 -> K 0..7/16..23 ; lanes>=16 -> 8..15/24..31
  const int nl = lane & 15;              // B/C: N = lane%16
  const int kb = (lane >> 4) * 8;        // B: lanes<16 -> K 0..7/16..23 ; lanes>=16 -> 8..15/24..31

  // staging addresses: thread tid -> tile row n=tid/4, k offset (tid%4)*8 (16B)
  const int sn = tid >> 2;
  const int sk = (tid & 3) * 8;
  const size_t brow = (size_t)(colB + sn) * K + sk;

  for (int k0 = 0; k0 < K; k0 += 32) {
    ((uint4*)Bs)[tid] = *(const uint4*)&B[brow + k0];
    __syncthreads();

    // A fragment: two 16B global loads per lane
    union { uint4 q[2]; v16bf v; } af;
    const u16* arow = A + (size_t)(row0 + m) * K + k0 + ka;
    af.q[0] = *(const uint4*)(arow);
    af.q[1] = *(const uint4*)(arow + 16);

#pragma unroll
    for (int c = 0; c < 4; ++c) {
      union { uint4 q[2]; v16bf v; } bf_;
      const uint4* bp = (const uint4*)&Bs[(c * 16 + nl) * 32 + kb];
      bf_.q[0] = bp[0];
      bf_.q[1] = bp[2];
      acc[c] = __builtin_amdgcn_wmma_f32_16x16x32_bf16(
          false, af.v, false, bf_.v, (short)0, acc[c], false, false);
    }
    __syncthreads();
  }

  // C layout: VGPR v -> M = v + 8*(lane>=16); N = lane%16
  const int mo = (lane >> 4) * 8;
#pragma unroll
  for (int c = 0; c < 4; ++c) {
    int col = colB + c * 16 + nl;
    float bv = bias[col];
#pragma unroll
    for (int v = 0; v < 8; ++v)
      C[(size_t)(row0 + mo + v) * N + col] = acc[c][v] + bv;
  }
}

// ---------- 4. persistent BiLSTM recurrence ----------
// grid = 16 WGs: dir = blk>>3 (0=fwd,1=bwd), 8 WGs per direction.
// Each WG owns 256 gate rows; W_hh slice (256x512 bf16) lives in LDS, loaded via TDM with
// pad_enable (1 dword pad per 512B row -> conflict-free column access).
__device__ __forceinline__ void dir_barrier(unsigned* arrive, unsigned* release,
                                            unsigned nwg, unsigned epoch) {
  __syncthreads();
#ifdef HAVE_CLUSTER_BAR
  __builtin_amdgcn_s_cluster_barrier();   // NOP outside a cluster launch; HW path if clustered
#endif
  if (threadIdx.x == 0) {
    __threadfence();
    unsigned total = atomicAdd(arrive, 1u) + 1u;
    if (total == epoch * nwg)
      __hip_atomic_store(release, epoch, __ATOMIC_RELEASE, __HIP_MEMORY_SCOPE_AGENT);
    while (__hip_atomic_load(release, __ATOMIC_ACQUIRE, __HIP_MEMORY_SCOPE_AGENT) < epoch)
      __builtin_amdgcn_s_sleep(1);
  }
  __syncthreads();
}

__global__ void lstm_recurrent_kernel(const u16* __restrict__ whh_bf16,  // [2][2048][512]
                                      const float* __restrict__ Gbuf,    // [2][L][2048]
                                      float* __restrict__ gates,         // [2][2048]
                                      float* __restrict__ h_cur,         // [2][512]
                                      float* __restrict__ c_cur,         // [2][512]
                                      u16* __restrict__ hcat,            // [L][1024]
                                      unsigned* __restrict__ sync) {
  extern __shared__ char smem[];
  // padded slice: 256 rows, row stride 516B (512B data + 4B pad) = 129 uints
  float* hbuf = (float*)(smem + 256 * 516);

  const int dir = blockIdx.x >> 3;
  const int wg  = blockIdx.x & 7;
  const u16* wsrc = whh_bf16 + (size_t)dir * G4 * H2 + (size_t)wg * 256 * H2;
  const float* G = Gbuf + (size_t)dir * LSEQ * G4;
  unsigned* arrive  = sync + dir * 2;
  unsigned* release = sync + dir * 2 + 1;

  // ---- preload W_hh slice into LDS ----
#ifdef HAVE_TDM
  if (threadIdx.x < 32) {
    unsigned long long ga = (unsigned long long)(const void*)wsrc;
    u32x4 g0 = { 1u,                                  // count=1
                 0u,                                  // lds_addr = 0 (dyn LDS base)
                 (unsigned)ga,
                 (unsigned)((ga >> 32) & 0x01FFFFFFu) | (2u << 30) };  // type=2
    i32x8 g1 = { (int)((1u << 16) | (1u << 20) | (6u << 22)),   // data_size=2B, pad_en, interval=128dw
                 (int)(512u << 16),                  // tensor_dim0 = 512
                 (int)(256u << 16),                  // tensor_dim1 = 256
                 (int)(512u << 16),                  // tile_dim0 = 512
                 256,                                // tile_dim1 = 256
                 512, 0, 0 };                        // dim0_stride = 512
    i32x4 z4 = {0, 0, 0, 0};
#if __clang_major__ >= 23
    i32x8 z8 = {0, 0, 0, 0, 0, 0, 0, 0};
    __builtin_amdgcn_tensor_load_to_lds(g0, g1, z4, z4, z8, 0);
#else
    __builtin_amdgcn_tensor_load_to_lds(g0, g1, z4, z4, 0);
#endif
    __builtin_amdgcn_s_wait_tensorcnt(0);
  }
#else
  {
    u16* wl = (u16*)smem;
    for (int i = threadIdx.x; i < 256 * 512; i += blockDim.x)
      wl[(i >> 9) * 258 + (i & 511)] = wsrc[i];
  }
#endif
  __syncthreads();

  const int r = threadIdx.x;             // 256 threads -> 256 gate rows
  const int R = wg * 256 + r;
  const unsigned* wrow = (const unsigned*)smem + r * 129;   // padded row
  unsigned epoch = 0;

  for (int step = 0; step < LSEQ; ++step) {
    const int pos = dir ? (LSEQ - 1 - step) : step;

    // stage h into LDS (vectorized: 128 lanes x float4 over first half of WG)
    if (threadIdx.x < H2 / 4)
      ((float4*)hbuf)[threadIdx.x] = ((const float4*)(h_cur + dir * H2))[threadIdx.x];
    __syncthreads();

    // gate row dot: 256 packed bf16 pairs against LDS-resident W_hh row
    const float2* h2p = (const float2*)hbuf;
    float sum = 0.f;
#pragma unroll 8
    for (int k2 = 0; k2 < 256; ++k2) {
      unsigned w = wrow[k2];
      float2 hv = h2p[k2];
      sum = fmaf(bf_lo(w), hv.x, sum);
      sum = fmaf(bf_hi(w), hv.y, sum);
    }
    gates[dir * G4 + R] = sum + G[(size_t)pos * G4 + R];

    // prefetch next step's gate-input row
    if (step + 1 < LSEQ) {
      int np = dir ? pos - 1 : pos + 1;
      __builtin_prefetch(&G[(size_t)np * G4 + R], 0, 1);
    }

    ++epoch; dir_barrier(arrive, release, 8u, epoch);

    // combine: this WG owns hidden units [wg*64, wg*64+64)
    if (threadIdx.x < 64) {
      int u = wg * 64 + threadIdx.x;
      float gi = gates[dir * G4 + u];
      float gf = gates[dir * G4 + H2 + u];
      float gg = gates[dir * G4 + 2 * H2 + u];
      float go = gates[dir * G4 + 3 * H2 + u];
      float c  = sigmoidf_(gf) * c_cur[dir * H2 + u] + sigmoidf_(gi) * tanhf(gg);
      float h  = sigmoidf_(go) * tanhf(c);
      c_cur[dir * H2 + u] = c;
      h_cur[dir * H2 + u] = h;
      hcat[(size_t)pos * (2 * H2) + dir * H2 + u] = f2bf(h);
    }

    ++epoch; dir_barrier(arrive, release, 8u, epoch);
  }
}

// ---------- 5. Viterbi forward (1 WG, trans in LDS) ----------
__global__ void viterbi_kernel(const float* __restrict__ emis, const float* __restrict__ trans,
                               const float* __restrict__ start_trans, const float* __restrict__ end_trans,
                               int* __restrict__ hist, float* __restrict__ sfinal) {
  extern __shared__ float sm[];
  float* tr = sm;                 // 128*128
  float* s0 = sm + NT * NT;       // 128
  float* s1 = s0 + NT;            // 128
  const int j = threadIdx.x;
  for (int i = j; i < NT * NT / 4; i += NT) ((float4*)tr)[i] = ((const float4*)trans)[i];
  s0[j] = start_trans[j] + emis[j];
  __syncthreads();
  for (int t = 1; t < LSEQ; ++t) {
    float* sp = (t & 1) ? s0 : s1;
    float* sn = (t & 1) ? s1 : s0;
    float best = -3.4e38f; int bi = 0;
#pragma unroll 4
    for (int i = 0; i < NT; ++i) {
      float v = sp[i] + tr[i * NT + j];
      if (v > best) { best = v; bi = i; }
    }
    sn[j] = best + emis[(size_t)t * NT + j];
    hist[(size_t)(t - 1) * NT + j] = bi;
    __syncthreads();
  }
  float* slast = ((LSEQ - 1) & 1) ? s1 : s0;
  sfinal[j] = slast[j] + end_trans[j];
}

// ---------- 6. backtrack ----------
__global__ void backtrack_kernel(const int* __restrict__ hist, const float* __restrict__ sfinal,
                                 float* __restrict__ out) {
  if (threadIdx.x != 0 || blockIdx.x != 0) return;
  float best = -3.4e38f; int last = 0;
  for (int i = 0; i < NT; ++i) { float v = sfinal[i]; if (v > best) { best = v; last = i; } }
  out[LSEQ - 1] = (float)last;
  for (int t = LSEQ - 2; t >= 0; --t) {
    last = hist[(size_t)t * NT + last];
    out[t] = (float)last;
  }
  out[LSEQ] = best;
}

// ---------- host ----------
extern "C" void kernel_launch(void* const* d_in, const int* in_sizes, int n_in,
                              void* d_out, int out_size, void* d_ws, size_t ws_size,
                              hipStream_t stream) {
  (void)in_sizes; (void)n_in; (void)out_size; (void)ws_size;
  const int*   sentence   = (const int*)d_in[0];
  const int*   intent     = (const int*)d_in[1];
  const float* word_emb   = (const float*)d_in[2];
  const float* intent_emb = (const float*)d_in[3];
  const float* w_ih_f     = (const float*)d_in[4];
  const float* w_hh_f     = (const float*)d_in[5];
  const float* b_f        = (const float*)d_in[6];
  const float* w_ih_b     = (const float*)d_in[7];
  const float* w_hh_b     = (const float*)d_in[8];
  const float* b_b        = (const float*)d_in[9];
  const float* h0         = (const float*)d_in[10];
  const float* c0         = (const float*)d_in[11];
  const float* lin_w      = (const float*)d_in[12];
  const float* lin_b      = (const float*)d_in[13];
  const float* trans      = (const float*)d_in[14];
  const float* start_tr   = (const float*)d_in[15];
  const float* end_tr     = (const float*)d_in[16];
  float* out = (float*)d_out;

  // workspace carve-up (256B aligned)
  char* ws = (char*)d_ws;
  size_t off = 0;
  auto alloc = [&](size_t bytes) { size_t r = off; off = (off + bytes + 255) & ~(size_t)255; return r; };
  u16*   x_bf16   = (u16*)(ws + alloc((size_t)LSEQ * DX * 2));
  u16*   wih_bf16 = (u16*)(ws + alloc((size_t)2 * G4 * DX * 2));       // [2][2048][640]
  u16*   whh_bf16 = (u16*)(ws + alloc((size_t)2 * G4 * H2 * 2));       // [2][2048][512]
  u16*   linw_b16 = (u16*)(ws + alloc((size_t)NT * (2 * H2) * 2));     // [128][1024]
  float* Gbuf     = (float*)(ws + alloc((size_t)2 * LSEQ * G4 * 4));   // [2][L][2048]
  float* gates    = (float*)(ws + alloc((size_t)2 * G4 * 4));
  float* h_cur    = (float*)(ws + alloc((size_t)2 * H2 * 4));
  float* c_cur    = (float*)(ws + alloc((size_t)2 * H2 * 4));
  u16*   hcat     = (u16*)(ws + alloc((size_t)LSEQ * (2 * H2) * 2));   // [L][1024]
  float* emis     = (float*)(ws + alloc((size_t)LSEQ * NT * 4));
  int*   hist     = (int*)(ws + alloc((size_t)(LSEQ - 1) * NT * 4));
  float* sfinal   = (float*)(ws + alloc((size_t)NT * 4));
  unsigned* sync  = (unsigned*)(ws + alloc(8 * 4));

  // 0. state init
  init_state_kernel<<<1, 1024, 0, stream>>>(h0, c0, h_cur, c_cur, sync);

  // 1. weight conversions to bf16 (vectorized by 4)
  auto cvt = [&](const float* src, u16* dst, int n) {
    int n4 = n / 4;
    int g = (n4 + 255) / 256; if (g > 2048) g = 2048;
    f32_to_bf16_kernel<<<g, 256, 0, stream>>>(src, dst, n4);
  };
  cvt(w_ih_f, wih_bf16,             G4 * DX);
  cvt(w_ih_b, wih_bf16 + G4 * DX,   G4 * DX);
  cvt(w_hh_f, whh_bf16,             G4 * H2);
  cvt(w_hh_b, whh_bf16 + G4 * H2,   G4 * H2);
  cvt(lin_w,  linw_b16,             NT * 2 * H2);

  // 2. embedding gather
  gather_embed_kernel<<<LSEQ, 128, 0, stream>>>(sentence, intent, word_emb, intent_emb, x_bf16);

  // 3. input projections (WMMA): G = x @ W_ih^T + b
  dim3 gproj(LSEQ / 128, G4 / 64);
  gemm_bf16_wmma_kernel<<<gproj, 256, 0, stream>>>(x_bf16, wih_bf16,            b_f,
                                                   Gbuf,                        LSEQ, G4, DX);
  gemm_bf16_wmma_kernel<<<gproj, 256, 0, stream>>>(x_bf16, wih_bf16 + G4 * DX,  b_b,
                                                   Gbuf + (size_t)LSEQ * G4,    LSEQ, G4, DX);

  // 4. recurrence: 16 WGs (8 per direction), 128KB+pad W_hh slice + h stage in LDS
  size_t lds_rec = 256 * 516 + H2 * 4;
  lstm_recurrent_kernel<<<16, 256, lds_rec, stream>>>(whh_bf16, Gbuf, gates, h_cur, c_cur, hcat, sync);

  // 5. emissions (WMMA): emis = hcat @ lin_w^T + lin_b
  dim3 gemi(LSEQ / 128, NT / 64);
  gemm_bf16_wmma_kernel<<<gemi, 256, 0, stream>>>(hcat, linw_b16, lin_b, emis, LSEQ, NT, 2 * H2);

  // 6. Viterbi
  size_t lds_vit = (size_t)(NT * NT + 2 * NT) * 4;
  viterbi_kernel<<<1, NT, lds_vit, stream>>>(emis, trans, start_tr, end_tr, hist, sfinal);

  // 7. backtrack -> d_out[0..L-1]=path, d_out[L]=best_score
  backtrack_kernel<<<1, 32, 0, stream>>>(hist, sfinal, out);
}
